// LoRALinear_12171937317054
// MI455X (gfx1250) — compile-verified, matrix operand order checked by reading, not run on hardware
//
#include <hip/hip_runtime.h>

// ---------------------------------------------------------------------------
// LoRA linear: out = x @ (W + A@B)^T + bias      (EFFECTIVE_SCALE == 1.0)
//   x:[M=8192,K=4096]f32  W:[N=4096,K]f32  bias:[N]f32  A:[N,16]f32  B:[16,K]f32
// Plan:
//   1) fold  W_eff = W + A@B          -> bf16   (ws)       ~0.5 GFLOP
//   2) cast  x                        -> bf16   (ws)       bandwidth-only
//   3) GEMM  out = x_bf16 @ W_eff^T + bias, v_wmma_f32_16x16x32_bf16,
//      double-buffered LDS fed by global_load_async_to_lds_b128 (ASYNCcnt).
// Compute-bound: 275 GFLOP bf16 vs ~15us of mandatory HBM traffic @23.3TB/s.
// ---------------------------------------------------------------------------

typedef __attribute__((ext_vector_type(16))) __bf16       v16bf;
typedef __attribute__((ext_vector_type(8)))  float        v8f;
typedef __attribute__((ext_vector_type(4)))  unsigned int u32x4;
typedef __attribute__((ext_vector_type(4)))  float        f32x4;

struct FragBits { u32x4 lo, hi; };   // 32 bytes == v16bf

__device__ __forceinline__ unsigned bfround(float f) {       // f32->bf16 RNE
  unsigned u = __builtin_bit_cast(unsigned, f);
  return (u + 0x7FFFu + ((u >> 16) & 1u)) >> 16;
}
__device__ __forceinline__ unsigned pack2(float lo, float hi) {
  return bfround(lo) | (bfround(hi) << 16);
}

// 16x32 bf16 WMMA operand fragment from LDS (ISA 7.12.2 layout):
// lanes 0-15: row=lane, K 0..7 + 16..23 ; lanes 16-31: row=lane-16, K 8..15 + 24..31.
__device__ __forceinline__ v16bf ld_frag(const unsigned short* p) {
  FragBits f;
  f.lo = *(const u32x4*)(p);
  f.hi = *(const u32x4*)(p + 16);
  return __builtin_bit_cast(v16bf, f);
}

// Async 16B global -> LDS copy, tracked by ASYNCcnt (CDNA5 §10 async ops).
// LDS byte address = low 32 bits of the generic shared-memory address
// (ISA §10.2 aperture).
__device__ __forceinline__ void async_b128(const unsigned short* g,
                                           unsigned short* l) {
  unsigned lds = (unsigned)(unsigned long long)l;
  unsigned long long ga = (unsigned long long)g;
  asm volatile("global_load_async_to_lds_b128 %0, %1, off"
               :: "v"(lds), "v"(ga) : "memory");
}
__device__ __forceinline__ void wait_async0() {
  asm volatile("s_wait_asynccnt 0x0" ::: "memory");
}

// ---------------------------------------------------------------------------
// Kernel 1: W_eff[n,k] = W[n,k] + sum_r A[n,r]*B[r,k]   (bf16 out)
// grid=(K/2048, N), 256 thr, 8 k/thread.
// ---------------------------------------------------------------------------
__global__ __launch_bounds__(256) void lora_fold_w(
    const float* __restrict__ W, const float* __restrict__ A,
    const float* __restrict__ Bm, unsigned short* __restrict__ weff, int K) {
  const int o = blockIdx.y;
  const int k = blockIdx.x * (256 * 8) + threadIdx.x * 8;

  float av[16];
#pragma unroll
  for (int r = 0; r < 16; ++r) av[r] = A[o * 16 + r];

  const float* wp = W + (size_t)o * K + k;
  f32x4 w0 = *(const f32x4*)(wp);
  f32x4 w1 = *(const f32x4*)(wp + 4);
  float acc[8] = {w0.x, w0.y, w0.z, w0.w, w1.x, w1.y, w1.z, w1.w};

#pragma unroll
  for (int r = 0; r < 16; ++r) {
    const float* bp = Bm + (size_t)r * K + k;
    f32x4 q0 = *(const f32x4*)(bp);
    f32x4 q1 = *(const f32x4*)(bp + 4);
    acc[0] = fmaf(av[r], q0.x, acc[0]);
    acc[1] = fmaf(av[r], q0.y, acc[1]);
    acc[2] = fmaf(av[r], q0.z, acc[2]);
    acc[3] = fmaf(av[r], q0.w, acc[3]);
    acc[4] = fmaf(av[r], q1.x, acc[4]);
    acc[5] = fmaf(av[r], q1.y, acc[5]);
    acc[6] = fmaf(av[r], q1.z, acc[6]);
    acc[7] = fmaf(av[r], q1.w, acc[7]);
  }
  u32x4 pk;
  pk.x = pack2(acc[0], acc[1]);
  pk.y = pack2(acc[2], acc[3]);
  pk.z = pack2(acc[4], acc[5]);
  pk.w = pack2(acc[6], acc[7]);
  *(u32x4*)(weff + (size_t)o * K + k) = pk;
}

// ---------------------------------------------------------------------------
// Kernel 2: x f32 -> bf16 (one-time; removes per-k-tile pack VALU in GEMM)
// ---------------------------------------------------------------------------
__global__ __launch_bounds__(256) void x_to_bf16(
    const float* __restrict__ x, unsigned short* __restrict__ xb) {
  const size_t i = ((size_t)blockIdx.x * 256 + threadIdx.x) * 8;
  f32x4 a0 = *(const f32x4*)(x + i);
  f32x4 a1 = *(const f32x4*)(x + i + 4);
  u32x4 pk;
  pk.x = pack2(a0.x, a0.y);
  pk.y = pack2(a0.z, a0.w);
  pk.z = pack2(a1.x, a1.y);
  pk.w = pack2(a1.z, a1.w);
  *(u32x4*)(xb + i) = pk;
}

// ---------------------------------------------------------------------------
// Kernel 3: main GEMM.  Block tile 128x256, BK=32, 256 thr = 8 wave32
// (2 along M x 4 along N), each wave a 64x64 tile = 4x4 WMMA accumulators.
// Double-buffered LDS filled by async b128 copies; per iter: issue next tile,
// 16 WMMAs on current, s_wait_asynccnt 0, one barrier.
// ---------------------------------------------------------------------------
#define BM 128
#define BN 256
#define BK 32
#define LDST 40   // padded LDS row stride (80B) -> conflict-free b128 frags

__global__ __launch_bounds__(256) void lora_gemm_async(
    const unsigned short* __restrict__ xb, const unsigned short* __restrict__ weff,
    const float* __restrict__ bias, float* __restrict__ out,
    int M, int N, int K) {
  __shared__ unsigned short Atile[2][BM * LDST];   // 2 x 10.0 KB
  __shared__ unsigned short Btile[2][BN * LDST];   // 2 x 20.0 KB

  const int tid  = threadIdx.x;
  const int wave = tid >> 5;
  const int lane = tid & 31;
  const int lrow = lane & 15;
  const int half = lane >> 4;

  const int bm = blockIdx.y * BM;
  const int bn = blockIdx.x * BN;
  const int wm = (wave & 1) * 64;
  const int wn = (wave >> 1) * 64;

  // Per-thread async copy slots: A = 2 x (row, 16B chunk), B = 4 x.
  auto issue = [&](int buf, int k0) {
#pragma unroll
    for (int c = 0; c < 2; ++c) {
      const int li = c * 256 + tid;
      const int r = li >> 2, q = (li & 3) * 8;
      async_b128(xb + (size_t)(bm + r) * K + k0 + q,
                 &Atile[buf][r * LDST + q]);
    }
#pragma unroll
    for (int c = 0; c < 4; ++c) {
      const int li = c * 256 + tid;
      const int r = li >> 2, q = (li & 3) * 8;
      async_b128(weff + (size_t)(bn + r) * K + k0 + q,
                 &Btile[buf][r * LDST + q]);
    }
  };

  v8f acc[4][4] = {};
  const int nk = K / BK;

  issue(0, 0);                                   // prologue fill
  for (int kt = 0; kt < nk; ++kt) {
    const int cur = kt & 1;
    wait_async0();        // my fills of buffer `cur` have landed
    __syncthreads();      // everyone's fills landed; prev reads of `cur^1` done
    if (kt + 1 < nk) issue(cur ^ 1, (kt + 1) * BK);   // overlap with compute

    v16bf afr[4], bfr[4];
#pragma unroll
    for (int i = 0; i < 4; ++i)
      afr[i] = ld_frag(&Atile[cur][(wm + i * 16 + lrow) * LDST + half * 8]);
#pragma unroll
    for (int j = 0; j < 4; ++j)
      bfr[j] = ld_frag(&Btile[cur][(wn + j * 16 + lrow) * LDST + half * 8]);

#pragma unroll
    for (int i = 0; i < 4; ++i)
#pragma unroll
      for (int j = 0; j < 4; ++j)
        acc[i][j] = __builtin_amdgcn_wmma_f32_16x16x32_bf16(
            false, afr[i], false, bfr[j], (short)0, acc[i][j], false, false);
  }

  // Epilogue: C layout -> VGPR r: M = half*8 + r, N = lane%16.
#pragma unroll
  for (int j = 0; j < 4; ++j) {
    const int gn = bn + wn + j * 16 + lrow;
    const float bv = bias[gn];
#pragma unroll
    for (int i = 0; i < 4; ++i) {
      const int gm0 = bm + wm + i * 16 + half * 8;
#pragma unroll
      for (int r = 0; r < 8; ++r)
        out[(size_t)(gm0 + r) * N + gn] = acc[i][j][r] + bv;
    }
  }
}

// ---------------------------------------------------------------------------
// Fallback GEMM (round-1): f32 x converted on the fly, 128x128 tile.
// Used only if ws is too small for the bf16 copy of x.
// ---------------------------------------------------------------------------
#define FLDST 40
__global__ __launch_bounds__(256) void lora_gemm_fb(
    const float* __restrict__ x, const unsigned short* __restrict__ weff,
    const float* __restrict__ bias, float* __restrict__ out,
    int M, int N, int K) {
  __shared__ unsigned short Asm_[128 * FLDST];
  __shared__ unsigned short Bsm_[128 * FLDST];

  const int tid  = threadIdx.x;
  const int wave = tid >> 5;
  const int lane = tid & 31;
  const int lrow = lane & 15;
  const int half = lane >> 4;
  const int bm = blockIdx.y * 128, bn = blockIdx.x * 128;
  const int wm = (wave & 1) * 64, wn = (wave >> 1) * 32;
  const int trow = tid >> 1, tcg = (tid & 1) * 16;

  const float*          xld = x    + (size_t)(bm + trow) * K + tcg;
  const unsigned short* wld = weff + (size_t)(bn + trow) * K + tcg;
  unsigned short* aSt = &Asm_[trow * FLDST + tcg];
  unsigned short* bSt = &Bsm_[trow * FLDST + tcg];

  v8f acc[4][2] = {};
  for (int k0 = 0; k0 < K; k0 += 32) {
    f32x4 a0 = *(const f32x4*)(xld + k0);
    f32x4 a1 = *(const f32x4*)(xld + k0 + 4);
    f32x4 a2 = *(const f32x4*)(xld + k0 + 8);
    f32x4 a3 = *(const f32x4*)(xld + k0 + 12);
    u32x4 b0 = *(const u32x4*)(wld + k0);
    u32x4 b1 = *(const u32x4*)(wld + k0 + 8);
    u32x4 p0, p1;
    p0.x = pack2(a0.x, a0.y); p0.y = pack2(a0.z, a0.w);
    p0.z = pack2(a1.x, a1.y); p0.w = pack2(a1.z, a1.w);
    p1.x = pack2(a2.x, a2.y); p1.y = pack2(a2.z, a2.w);
    p1.z = pack2(a3.x, a3.y); p1.w = pack2(a3.z, a3.w);
    __syncthreads();
    *(u32x4*)aSt = p0; *(u32x4*)(aSt + 8) = p1;
    *(u32x4*)bSt = b0; *(u32x4*)(bSt + 8) = b1;
    __syncthreads();
    v16bf afr[4], bfr[2];
#pragma unroll
    for (int i = 0; i < 4; ++i)
      afr[i] = ld_frag(&Asm_[(wm + i * 16 + lrow) * FLDST + half * 8]);
#pragma unroll
    for (int j = 0; j < 2; ++j)
      bfr[j] = ld_frag(&Bsm_[(wn + j * 16 + lrow) * FLDST + half * 8]);
#pragma unroll
    for (int i = 0; i < 4; ++i)
#pragma unroll
      for (int j = 0; j < 2; ++j)
        acc[i][j] = __builtin_amdgcn_wmma_f32_16x16x32_bf16(
            false, afr[i], false, bfr[j], (short)0, acc[i][j], false, false);
  }
#pragma unroll
  for (int j = 0; j < 2; ++j) {
    const int gn = bn + wn + j * 16 + lrow;
    const float bv = bias[gn];
#pragma unroll
    for (int i = 0; i < 4; ++i) {
      const int gm0 = bm + wm + i * 16 + half * 8;
#pragma unroll
      for (int r = 0; r < 8; ++r)
        out[(size_t)(gm0 + r) * N + gn] = acc[i][j][r] + bv;
    }
  }
}

// ---------------------------------------------------------------------------
extern "C" void kernel_launch(void* const* d_in, const int* in_sizes, int n_in,
                              void* d_out, int out_size, void* d_ws, size_t ws_size,
                              hipStream_t stream) {
  const float* x    = (const float*)d_in[0];
  const float* W    = (const float*)d_in[1];
  const float* bias = (const float*)d_in[2];
  const float* A    = (const float*)d_in[3];
  const float* Bm   = (const float*)d_in[4];
  float* out = (float*)d_out;

  const int N = in_sizes[2];        // OUT (bias length)      = 4096
  const int K = in_sizes[4] / 16;   // IN  (B is [16, IN])    = 4096
  const int M = in_sizes[0] / K;    // B*S                    = 8192
  (void)n_in; (void)out_size;

  unsigned short* weff = (unsigned short*)d_ws;            // [N,K] bf16
  const size_t weff_b = (size_t)N * K * sizeof(unsigned short);
  const size_t xb_b   = (size_t)M * K * sizeof(unsigned short);

  lora_fold_w<<<dim3(K / 2048, N), 256, 0, stream>>>(W, A, Bm, weff, K);

  if (ws_size >= weff_b + xb_b) {
    unsigned short* xb = (unsigned short*)((char*)d_ws + weff_b);
    x_to_bf16<<<dim3((unsigned)(((size_t)M * K) / 2048)), 256, 0, stream>>>(x, xb);
    lora_gemm_async<<<dim3(N / BN, M / BM), 256, 0, stream>>>(xb, weff, bias,
                                                              out, M, N, K);
  } else {
    lora_gemm_fb<<<dim3(N / 128, M / 128), 256, 0, stream>>>(x, weff, bias,
                                                             out, M, N, K);
  }
}
